// Convolution_1176821039998
// MI455X (gfx1250) — matrix operand order repaired and checked by reading, not compiled, not run on hardware
//
#include <hip/hip_runtime.h>

typedef __attribute__((ext_vector_type(16))) __bf16 v16bf;
typedef __attribute__((ext_vector_type(8)))  __bf16 v8bf;
typedef __attribute__((ext_vector_type(8)))  float  v8f;
typedef __attribute__((ext_vector_type(4)))  unsigned int u32x4;
typedef __attribute__((ext_vector_type(8)))  int i32x8;
typedef __attribute__((ext_vector_type(4)))  int i32x4;

// ---------------------------------------------------------------------------
// Kernel 1: build conv weights [tap=125][co=64][ci=64] in bf16 from `weight`,
// folding the skip-connection (0.25*w_sc0 / 0.25*w_sc1*I3) into center tap 62.
// ---------------------------------------------------------------------------
__device__ __forceinline__ float sus_f(float t) {
    return t > 0.f ? __expf(-1.f / t) : 0.f;
}

__global__ void build_wk_kernel(const float* __restrict__ weight,
                                const float* __restrict__ w_sc0,
                                const float* __restrict__ w_sc1,
                                __bf16* __restrict__ wk) {
    const int l = blockIdx.x;                    // tap 0..124  (a,b,c)
    const int a = l / 25, b = (l / 5) % 5, c = l % 5;
    const float lx = (a - 2) * 0.5f, ly = (b - 2) * 0.5f, lz = (c - 2) * 0.5f;
    const float d  = sqrtf(lx * lx + ly * ly + lz * lz);
    const float dn = fmaxf(d, 1e-12f);
    const float SQ3 = 1.7320508075688772f;
    const float sh1[3] = { SQ3 * lx / dn, SQ3 * ly / dn, SQ3 * lz / dn };

    float emb[5];
    const float C = 1.14136f * 7.3890560989306495f;   // 1.14136 * e^2
    #pragma unroll
    for (int s = 0; s < 5; ++s) {
        float diff = 4.f * d - (float)s;              // (d - s*step)/step
        emb[s] = C * sus_f(diff + 1.f) * sus_f(1.f - diff);
    }
    const float PW0 = 0.17677669529663687f;           // sqrt(1/32)
    const float PW1 = 0.30618621784789724f;           // sqrt(3/32)
    const float IS3 = 0.57735026918962576f;           // 1/sqrt(3)

    for (int e = threadIdx.x; e < 64 * 64; e += blockDim.x) {
        const int col = e >> 6;                       // out channel o
        const int row = e & 63;                       // in  channel i
        float val;
        if (row < 16 && col < 16) {                   // k_ss : w1 block
            const int cidx = row * 16 + col;
            float w = 0.f;
            #pragma unroll
            for (int s = 0; s < 5; ++s) w += emb[s] * weight[s * 1024 + cidx];
            val = PW0 * w;
            if (l == 62) val += 0.25f * w_sc0[row * 16 + col];
        } else if (row < 16) {                        // k_sv : w2 block
            const int v = (col - 16) / 3, k = (col - 16) % 3;
            const int cidx = 256 + row * 16 + v;
            float w = 0.f;
            #pragma unroll
            for (int s = 0; s < 5; ++s) w += emb[s] * weight[s * 1024 + cidx];
            val = PW1 * IS3 * w * sh1[k];
        } else if (col < 16) {                        // k_vs : w4 block
            const int u = (row - 16) / 3, i = (row - 16) % 3;
            const int cidx = 768 + u * 16 + col;
            float w = 0.f;
            #pragma unroll
            for (int s = 0; s < 5; ++s) w += emb[s] * weight[s * 1024 + cidx];
            val = PW0 * IS3 * w * sh1[i];
        } else {                                      // k_vv : w3 block * I3
            const int u = (row - 16) / 3, i = (row - 16) % 3;
            const int v = (col - 16) / 3, j = (col - 16) % 3;
            if (i == j) {
                const int cidx = 512 + u * 16 + v;
                float w = 0.f;
                #pragma unroll
                for (int s = 0; s < 5; ++s) w += emb[s] * weight[s * 1024 + cidx];
                val = PW1 * IS3 * w;
                if (l == 62) val += 0.25f * w_sc1[u * 16 + v];
            } else {
                val = 0.f;
            }
        }
        wk[((size_t)l * 64 + col) * 64 + row] = (__bf16)val;   // [tap][co][ci]
    }
}

// ---------------------------------------------------------------------------
// Kernel 2: one-time transpose + fp32->bf16: xt[z][y][x][ci] (ci fastest).
// Makes all inner-loop staging pure contiguous copies -> TDM-friendly.
// ---------------------------------------------------------------------------
__global__ void __launch_bounds__(256)
xt_transpose_kernel(const float* __restrict__ x, __bf16* __restrict__ xt) {
    __shared__ __bf16 t[64][72];                 // padded rows (144 B)
    const int tid = threadIdx.x;
    const int v0  = blockIdx.x * 64;             // 2744 * 64 = 56^3 exactly
    #pragma unroll
    for (int c = 0; c < 16; ++c) {               // coalesced read over voxels
        const int ci = c * 4 + (tid >> 6);
        const int vi = tid & 63;
        t[vi][ci] = (__bf16)x[(size_t)ci * 175616 + v0 + vi];
    }
    __syncthreads();
    #pragma unroll
    for (int c = 0; c < 16; ++c) {               // coalesced write over ci
        const int vi = c * 4 + (tid >> 6);
        const int ci = tid & 63;
        xt[(size_t)(v0 + vi) * 64 + ci] = t[vi][ci];
    }
}

// ---------------------------------------------------------------------------
// TDM helper: 1-D tile load (tile0 bf16 elements) global -> LDS, optional
// per-128B LDS padding of 16 B (rows of 64 halfs -> 72 halfs).
// D# per CDNA5 ISA ch.8: group0 {count=1, lds_addr, global_addr, type=2},
// group1 {data_size=2B, pad ctl, tensor_dim0/1, tile_dim0/1, dim0 stride}.
// This toolchain's builtin takes 6 args (groups 2/3 + trailing group, cpol).
// ---------------------------------------------------------------------------
__device__ __forceinline__ void tdm_load_1d(const void* lds_dst,
                                            const void* gsrc,
                                            unsigned tile0, unsigned td0,
                                            int pad_en) {
    const unsigned lds_off = (unsigned)(unsigned long long)lds_dst;
    const unsigned long long ga = (unsigned long long)gsrc;
    u32x4 g0;
    g0[0] = 1u;                                   // count=1 (valid user D#)
    g0[1] = lds_off;                              // lds_addr (bytes)
    g0[2] = (unsigned)ga;                         // global_addr[31:0]
    g0[3] = (unsigned)(ga >> 32) | (2u << 30);    // global_addr[56:32] | type=2
    i32x8 g1;
    g1[0] = (1 << 16)                             // data_size = 1 -> 2 bytes
          | (pad_en ? ((1 << 20)                  // pad_enable
                     | (4 << 22)                  // pad_interval: 32 DWORDs
                     | (3 << 25)) : 0);           // pad_amount:   4 DWORDs
    g1[1] = (int)((td0 & 0xFFFFu) << 16);         // tensor_dim0[15:0]
    g1[2] = (int)((td0 >> 16) & 0xFFFFu)          // tensor_dim0[31:16]
          | (1 << 16);                            // tensor_dim1 = 1
    g1[3] = (int)((tile0 & 0xFFFFu) << 16);       // tile_dim0
    g1[4] = 1;                                    // tile_dim1 = 1, tile_dim2 = 0
    g1[5] = (int)td0;                             // tensor_dim0_stride[31:0]
    g1[6] = 0;
    g1[7] = 0;
    const i32x4 z4 = {0, 0, 0, 0};                // groups 2/3 unused (<=2D)
    const i32x8 z8 = {0, 0, 0, 0, 0, 0, 0, 0};
    __builtin_amdgcn_tensor_load_to_lds(g0, g1, z4, z4, z8, 0);
}

// ---------------------------------------------------------------------------
// Kernel 3 (fast path): implicit-GEMM conv, TDM double-buffered staging.
// Workgroup = 64 co x 64 voxels (full x row at fixed y0,z0); 8 waves; each
// wave owns 2 D tiles (16 co x 16 x) sharing one A fragment.
// ---------------------------------------------------------------------------
__global__ void __launch_bounds__(256)
conv_wmma_tdm_kernel(const __bf16* __restrict__ xt,
                     const __bf16* __restrict__ wk,
                     float* __restrict__ out) {
    __shared__ __bf16 xbuf[2][56][72];            // [buf][x 0..55][ci], 144 B rows
    __shared__ __bf16 wbuf[2][5][64][64];         // [buf][dx][co][ci]

    const int tid  = threadIdx.x;
    const int lane = tid & 31;                    // wave32
    const int wave = tid >> 5;
    const int co_tile = wave & 3;                 // 4 co tiles of 16
    const int npair   = wave >> 2;                // voxel pair-half: 0 or 1

    const int y0 = blockIdx.x;
    const int z0 = blockIdx.y;

    const int n     = lane & 15;
    const int hi    = lane >> 4;
    const int koff  = hi * 16;                    // B K offset
    const int kbase = hi * 16;                    // A K offset
    const int co_l  = co_tile * 16 + n;           // A row (per-lane co)

    v8f acc0 = {}, acc1 = {};

    // ---- prologue: TDM-stage iteration 0 into buffer 0 ----
    if (wave == 0) {
        tdm_load_1d(&xbuf[0][0][0],
                    xt + (((size_t)z0 * 56 + y0) * 56) * 64,
                    3584u, 11239424u, 1);
        tdm_load_1d(&wbuf[0][0][0][0], wk, 20480u, 512000u, 0);
        __builtin_amdgcn_s_wait_tensorcnt(0);
    }
    __syncthreads();

    #pragma unroll 1
    for (int it = 0; it < 25; ++it) {
        const int cur = it & 1, nxt = cur ^ 1;

        // ---- issue async TDM loads for iteration it+1 ----
        if (it + 1 < 25 && wave == 0) {
            const int dz2 = (it + 1) / 5, dy2 = (it + 1) % 5;
            tdm_load_1d(&xbuf[nxt][0][0],
                        xt + (((size_t)(z0 + dz2) * 56 + (y0 + dy2)) * 56) * 64,
                        3584u, 11239424u, 1);
            tdm_load_1d(&wbuf[nxt][0][0][0],
                        wk + (size_t)((dz2 * 5 + dy2) * 5) * 4096,
                        20480u, 512000u, 0);
        }

        // ---- compute on current buffers: 5 dx * 2 K-chunks * 2 tiles ----
        #pragma unroll
        for (int dx = 0; dx < 5; ++dx) {
            #pragma unroll
            for (int ch = 0; ch < 2; ++ch) {
                const int c32 = ch * 32;
                union { v16bf v; v8bf h[2]; } afrag;
                afrag.h[0] = *(const v8bf*)&wbuf[cur][dx][co_l][c32 + kbase];
                afrag.h[1] = *(const v8bf*)&wbuf[cur][dx][co_l][c32 + kbase + 8];
                const int xiA = dx + npair * 32 + n;        // tile 0 row
                const int xiB = min(xiA + 16, 55);          // tile 1 row (clamped)
                const v16bf b0 = *(const v16bf*)&xbuf[cur][xiA][c32 + koff];
                const v16bf b1 = *(const v16bf*)&xbuf[cur][xiB][c32 + koff];
                acc0 = __builtin_amdgcn_wmma_f32_16x16x32_bf16(
                    false, afrag.v, false, b0, (short)0, acc0, false, false);
                acc1 = __builtin_amdgcn_wmma_f32_16x16x32_bf16(
                    false, afrag.v, false, b1, (short)0, acc1, false, false);
            }
        }

        // ---- wait for next-iter DMA, then release buffers ----
        if (wave == 0) __builtin_amdgcn_s_wait_tensorcnt(0);
        __syncthreads();
    }

    // ---- store: lane col = x voxel; VGPR r -> co row ----
    const int xv0 = npair * 32 + n;
    const int xv1 = xv0 + 16;
    #pragma unroll
    for (int r = 0; r < 8; ++r) {
        const int o = co_tile * 16 + hi * 8 + r;
        const size_t base = ((size_t)o * 52 + z0) * 52 + y0;
        if (xv0 < 52) out[base * 52 + xv0] = acc0[r];
        if (xv1 < 52) out[base * 52 + xv1] = acc1[r];
    }
}

// ---------------------------------------------------------------------------
// Kernel 3 (fallback, small workspace): round-1 manual-staging conv.
// ---------------------------------------------------------------------------
__global__ void __launch_bounds__(256)
conv_wmma_kernel(const float* __restrict__ x,
                 const __bf16* __restrict__ wk,
                 float* __restrict__ out) {
    __shared__ __bf16 xbuf[36][72];
    __shared__ __bf16 wbuf[5][64][64];

    const int tid  = threadIdx.x;
    const int lane = tid & 31;
    const int wave = tid >> 5;
    const int co_tile = wave & 3;
    const int vhalf   = wave >> 2;

    const int x0 = blockIdx.x * 32;
    const int y0 = blockIdx.y;
    const int z0 = blockIdx.z;

    const int n     = lane & 15;
    const int hi    = lane >> 4;
    const int koff  = hi * 16;
    const int kbase = hi * 16;
    const int co_l  = co_tile * 16 + n;

    v8f acc = {};

    #pragma unroll 1
    for (int dz = 0; dz < 5; ++dz) {
        #pragma unroll 1
        for (int dy = 0; dy < 5; ++dy) {
            __syncthreads();
            {
                const int zg = z0 + dz, yg = y0 + dy;
                #pragma unroll
                for (int it = 0; it < 9; ++it) {
                    const int i  = tid + it * 256;
                    const int ci = i / 36;
                    const int xi = i - ci * 36;
                    const int xg = (x0 + xi < 56) ? (x0 + xi) : 55;
                    const float v = x[(((size_t)ci * 56 + zg) * 56 + yg) * 56 + xg];
                    xbuf[xi][ci] = (__bf16)v;
                }
            }
            {
                const int tap0 = (dz * 5 + dy) * 5;
                const v8bf* src = (const v8bf*)(wk + (size_t)tap0 * 4096);
                v8bf*       dst = (v8bf*)&wbuf[0][0][0];
                #pragma unroll
                for (int it = 0; it < 10; ++it)
                    dst[tid + it * 256] = src[tid + it * 256];
            }
            __syncthreads();

            #pragma unroll
            for (int dx = 0; dx < 5; ++dx) {
                #pragma unroll
                for (int ch = 0; ch < 2; ++ch) {
                    const int c32 = ch * 32;
                    union { v16bf v; v8bf h[2]; } afrag;
                    afrag.h[0] = *(const v8bf*)&wbuf[dx][co_l][c32 + kbase];
                    afrag.h[1] = *(const v8bf*)&wbuf[dx][co_l][c32 + kbase + 8];
                    const v16bf bfrag =
                        *(const v16bf*)&xbuf[dx + vhalf * 16 + n][c32 + koff];
                    acc = __builtin_amdgcn_wmma_f32_16x16x32_bf16(
                        false, afrag.v, false, bfrag, (short)0, acc,
                        false, false);
                }
            }
        }
    }

    const int xv = x0 + vhalf * 16 + n;
    if (xv < 52) {
        #pragma unroll
        for (int r = 0; r < 8; ++r) {
            const int o = co_tile * 16 + hi * 8 + r;
            out[(((size_t)o * 52 + z0) * 52 + y0) * 52 + xv] = acc[r];
        }
    }
}

// ---------------------------------------------------------------------------
extern "C" void kernel_launch(void* const* d_in, const int* in_sizes, int n_in,
                              void* d_out, int out_size, void* d_ws, size_t ws_size,
                              hipStream_t stream) {
    (void)in_sizes; (void)n_in; (void)out_size;
    const float* x      = (const float*)d_in[0];
    const float* weight = (const float*)d_in[1];
    const float* wsc0   = (const float*)d_in[2];
    const float* wsc1   = (const float*)d_in[3];
    float*  out = (float*)d_out;

    __bf16* wk = (__bf16*)d_ws;                         // 1 MB region
    __bf16* xt = (__bf16*)((char*)d_ws + (1u << 20));   // 22.5 MB region
    const size_t need = (1ull << 20) + 56ull * 56 * 56 * 64 * 2;

    build_wk_kernel<<<125, 256, 0, stream>>>(weight, wsc0, wsc1, wk);
    if (ws_size >= need) {
        xt_transpose_kernel<<<2744, 256, 0, stream>>>(x, xt);
        conv_wmma_tdm_kernel<<<dim3(52, 52), 256, 0, stream>>>(xt, wk, out);
    } else {
        conv_wmma_kernel<<<dim3(2, 52, 52), 256, 0, stream>>>(x, wk, out);
    }
}
